// DigitCapsule_34686155883103
// MI455X (gfx1250) — compile-verified, matrix operand order checked by reading, not run on hardware
//
#include <hip/hip_runtime.h>

#define NB 64
#define NN 4096
#define NC 10
#define ND 8
#define NE 16
#define CE (NC * NE)          // 160 floats per (b,n) row of u_hat
#define CAP_EPS 1e-7f

typedef __attribute__((ext_vector_type(16))) _Float16 v16h;
typedef __attribute__((ext_vector_type(8)))  float    v8f;

// ---------------------------------------------------------------- zero fill
__global__ void zero_k(float* __restrict__ p, int n) {
  int i = blockIdx.x * blockDim.x + threadIdx.x;
  if (i < n) p[i] = 0.f;
}

// ---------------------------------------------------------------- u_hat GEMM (WMMA)
// u_hat[b,n,c,e] = sum_d x[b,n,d] * W[n,c,d,e]
// One block per n; 4 waves; wave w owns batch rows 16w..16w+15 (M tile).
// f32 accuracy from a single f16 WMMA via hi/lo split packed into K slots:
//   K0..7 : x_hi*w_hi    K8..15 : x_hi*w_lo    K16..23 : x_lo*w_hi
__global__ __launch_bounds__(128) void uhat_gemm_k(const float* __restrict__ x,
                                                   const float* __restrict__ W,
                                                   float* __restrict__ uhat) {
  __shared__ float wt[NC * NE * ND];   // W[n] transposed to [c][e][d], 5 KB

  const int n    = blockIdx.x;
  const int lane = threadIdx.x & 31;
  const int wave = threadIdx.x >> 5;
  const int sub  = lane & 15;          // A row / B col / D col (= e)
  const bool hih = lane >= 16;

  // ---- stage W[n] (1280 contiguous floats) into LDS, transposed
  {
    const float4* wsrc = (const float4*)(W + (size_t)n * (NC * ND * NE));
    for (int m4 = threadIdx.x; m4 < (NC * ND * NE) / 4; m4 += 128) {
      float4 w4 = wsrc[m4];
      int m = m4 * 4;                 // linear [c][d][e], e fastest
      int c = m / (ND * NE);
      int r = m - c * (ND * NE);
      int d = r / NE;
      int e = r - d * NE;             // e, e+1, e+2, e+3 within the float4
      float* dst = &wt[(c * NE + e) * ND + d];
      dst[0 * ND] = w4.x;
      dst[1 * ND] = w4.y;
      dst[2 * ND] = w4.z;
      dst[3 * ND] = w4.w;
    }
  }

  // ---- A fragment: rows b = wave*16 + sub, d = 0..7 (two b128 loads)
  const float* xp = x + ((size_t)(wave * 16 + sub) * NN + n) * ND;
  float4 xa = *(const float4*)xp;
  float4 xb = *(const float4*)(xp + 4);
  float xr[8] = {xa.x, xa.y, xa.z, xa.w, xb.x, xb.y, xb.z, xb.w};

  v16h a;
  #pragma unroll
  for (int d = 0; d < ND; ++d) {
    _Float16 h  = (_Float16)xr[d];
    _Float16 lo = (_Float16)(xr[d] - (float)h);
    a[d]     = h;                                // pairs w_hi (lo half) / w_lo (hi half)
    a[8 + d] = hih ? (_Float16)0.f : lo;         // pairs w_hi
  }

  __syncthreads();

  #pragma unroll
  for (int c = 0; c < NC; ++c) {
    // ---- B fragment from LDS: col e = sub, 8 contiguous d-values
    const float4* wp = (const float4*)&wt[(c * NE + sub) * ND];
    float4 w0 = wp[0], w1 = wp[1];
    float wr[8] = {w0.x, w0.y, w0.z, w0.w, w1.x, w1.y, w1.z, w1.w};

    v16h bf;
    #pragma unroll
    for (int d = 0; d < ND; ++d) {
      _Float16 h  = (_Float16)wr[d];
      _Float16 lo = (_Float16)(wr[d] - (float)h);
      bf[d]     = hih ? lo : h;                  // K0..7 = w_hi | K8..15 = w_lo
      bf[8 + d] = hih ? (_Float16)0.f : h;       // K16..23 = w_hi | K24..31 = 0
    }

    v8f acc = {};
    acc = __builtin_amdgcn_wmma_f32_16x16x32_f16(
        false, a, false, bf, (short)0, acc, false, false);

    // D layout: VGPR r, lane l -> M = r + 8*(l>=16), N = l&15
    const int b0 = wave * 16 + (hih ? 8 : 0);
    #pragma unroll
    for (int r = 0; r < 8; ++r) {
      uhat[(((size_t)(b0 + r) * NN + n) * NC + c) * NE + sub] = acc[r];
    }
  }
}

// ---------------------------------------------------------------- iter0: s = 0.1 * sum_n u_hat
#define SR_CHUNK 256
__global__ __launch_bounds__(160) void sum0_k(const float* __restrict__ uhat,
                                              float* __restrict__ s) {
  const int b = blockIdx.x;           // 0..63
  const int chunk = blockIdx.y;       // 0..15
  const int tid = threadIdx.x;        // (c,e)
  const size_t bn0 = (size_t)b * NN + chunk * SR_CHUNK;
  float acc = 0.f;
  for (int k = 0; k < SR_CHUNK; ++k)
    acc += uhat[(bn0 + k) * CE + tid];
  atomicAdd(&s[b * CE + tid], acc * (1.0f / NC));
}

// ---------------------------------------------------------------- squash
__global__ __launch_bounds__(160) void squash_k(const float* __restrict__ s,
                                                float* __restrict__ v) {
  const int b = blockIdx.x;
  const int c = threadIdx.x >> 4;
  __shared__ float red[CE];
  float sv = s[b * CE + threadIdx.x];
  red[threadIdx.x] = sv * sv;
  __syncthreads();
  float s2 = 0.f;
  #pragma unroll
  for (int i = 0; i < NE; ++i) s2 += red[c * NE + i];
  float scale = s2 / ((1.f + s2) * sqrtf(s2 + CAP_EPS));
  v[b * CE + threadIdx.x] = sv * scale;
}

// ---------------------------------------------------------------- fused routing step:
// blog += <u_hat, v>;  c = softmax(blog);  s += c * u_hat   (one u_hat pass)
#define RT_TILE 8
__global__ __launch_bounds__(160) void route_fused_k(const float* __restrict__ uhat,
                                                     const float* __restrict__ v,
                                                     float* __restrict__ blog,
                                                     float* __restrict__ s) {
  const int b = blockIdx.x;           // 0..63
  const int chunk = blockIdx.y;       // 0..15
  const int tid = threadIdx.x;
  const int c = tid >> 4;
  const int e = tid & 15;
  __shared__ float dotl[RT_TILE][NC];

  const float vv = v[b * CE + tid];
  const size_t bn0 = (size_t)b * NN + chunk * SR_CHUNK;
  float sacc = 0.f;

  for (int kk = 0; kk < SR_CHUNK; kk += RT_TILE) {
    float uarr[RT_TILE];
    #pragma unroll
    for (int j = 0; j < RT_TILE; ++j) {
      const size_t bn = bn0 + kk + j;
      float u = uhat[bn * CE + tid];
      uarr[j] = u;
      float t = u * vv;
      // reduce over e (16 lanes) within the wave
      #pragma unroll
      for (int off = 8; off; off >>= 1) t += __shfl_xor(t, off, 16);
      if (e == 0) {
        float nb = blog[bn * NC + c] + t;
        blog[bn * NC + c] = nb;
        dotl[j][c] = nb;
      }
    }
    __syncthreads();
    #pragma unroll
    for (int j = 0; j < RT_TILE; ++j) {
      float m = dotl[j][0];
      #pragma unroll
      for (int i = 1; i < NC; ++i) m = fmaxf(m, dotl[j][i]);
      float sum = 0.f;
      #pragma unroll
      for (int i = 0; i < NC; ++i) sum += __expf(dotl[j][i] - m);
      float w = __expf(dotl[j][c] - m) / sum;
      sacc += w * uarr[j];
    }
    __syncthreads();
  }
  atomicAdd(&s[b * CE + tid], sacc);
}

// ---------------------------------------------------------------- launch
extern "C" void kernel_launch(void* const* d_in, const int* in_sizes, int n_in,
                              void* d_out, int out_size, void* d_ws, size_t ws_size,
                              hipStream_t stream) {
  (void)in_sizes; (void)n_in; (void)out_size; (void)ws_size;
  const float* x = (const float*)d_in[0];
  const float* W = (const float*)d_in[1];
  float* out = (float*)d_out;

  float* ws   = (float*)d_ws;
  float* uhat = ws;                                 // 64*4096*10*16
  float* blog = uhat + (size_t)NB * NN * CE;        // 64*4096*10
  float* s    = blog + (size_t)NB * NN * NC;        // 64*10*16
  float* v    = s + NB * CE;                        // 64*10*16

  const dim3 rgrid(NB, NN / SR_CHUNK);

  zero_k<<<(NB * NN * NC + 255) / 256, 256, 0, stream>>>(blog, NB * NN * NC);
  zero_k<<<(NB * CE + 255) / 256, 256, 0, stream>>>(s, NB * CE);
  uhat_gemm_k<<<NN, 128, 0, stream>>>(x, W, uhat);

  // iteration 0: uniform coupling (softmax of zeros) folded into a plain reduce
  sum0_k<<<rgrid, CE, 0, stream>>>(uhat, s);
  squash_k<<<NB, CE, 0, stream>>>(s, v);

  // iterations 1..2: fused agreement + logit update + softmax + s reduce
  for (int it = 1; it < 3; ++it) {
    zero_k<<<(NB * CE + 255) / 256, 256, 0, stream>>>(s, NB * CE);
    route_fused_k<<<rgrid, CE, 0, stream>>>(uhat, v, blog, s);
    squash_k<<<NB, CE, 0, stream>>>(s, (it == 2) ? out : v);
  }
}
